// Decoder_18270790877397
// MI455X (gfx1250) — compile-verified, matrix operand order checked by reading, not run on hardware
//
#include <hip/hip_runtime.h>
#include <hip/hip_bf16.h>

// ---------------------------------------------------------------------------
// Types for gfx1250 (wave32) WMMA + TDM
// ---------------------------------------------------------------------------
typedef __attribute__((ext_vector_type(16))) __bf16 v16bf;
typedef __attribute__((ext_vector_type(8)))  __bf16 v8bf;
typedef __attribute__((ext_vector_type(8)))  float  v8f;
typedef __attribute__((ext_vector_type(4)))  unsigned int v4u;
typedef __attribute__((ext_vector_type(8)))  unsigned int v8u;
typedef __attribute__((ext_vector_type(8)))  unsigned short v8us;

#define D_MODEL 2048
#define D_FF    8192
#define NHEAD   16
#define DKH     128
#define BATCHN  4
#define SEQN    2048
#define MTOT    (BATCHN * SEQN)   // 8192 rows

// ---------------------------------------------------------------------------
// fp32 -> bf16 (round to nearest even)
// ---------------------------------------------------------------------------
static __device__ __forceinline__ unsigned short f2bf_bits(float f) {
  unsigned int u = __builtin_bit_cast(unsigned int, f);
  u += 0x7FFFu + ((u >> 16) & 1u);
  return (unsigned short)(u >> 16);
}

static __device__ __forceinline__ v8f wmma_bf16(v16bf a, v16bf b, v8f c) {
  return __builtin_amdgcn_wmma_f32_16x16x32_bf16(
      false, a, false, b, (short)0, c, false, false);
}

// Fragment load (16 bf16) from k-contiguous memory: elements 0..7 from
// p[0..7], 8..15 from p[16..23]. p must be 16B aligned.
static __device__ __forceinline__ v16bf load16(const unsigned short* p) {
  v8bf lo = *(const v8bf*)(p + 0);
  v8bf hi = *(const v8bf*)(p + 16);
  return __builtin_shufflevector(lo, hi, 0,1,2,3,4,5,6,7,8,9,10,11,12,13,14,15);
}

// ---------------------------------------------------------------------------
// Tensor Data Mover: async 2D bf16 tile global -> LDS, with LDS padding of
// +4 DWORDs every 16 DWORDs (row 64B -> 80B stride). Descriptor per ISA ch.8.
// ---------------------------------------------------------------------------
static __device__ __forceinline__ void tdm_load_2d(
    unsigned lds_off, const void* gaddr,
    unsigned tensor_d0, unsigned tensor_d1,
    unsigned tile_d0, unsigned tile_d1, unsigned stride0) {
  unsigned long long ga = (unsigned long long)(uintptr_t)gaddr;
  v4u g0;
  g0[0] = 1u;                                  // count=1, user descriptor
  g0[1] = lds_off;                             // lds_addr (bytes)
  g0[2] = (unsigned)(ga & 0xFFFFFFFFu);        // global_addr[31:0]
  g0[3] = (unsigned)((ga >> 32) & 0x01FFFFFFu) // global_addr[56:32]
          | (2u << 30);                        // type=2 ("image")
  v8u g1;
  g1[0] = (1u << 16)    // data_size = 1 (2 bytes, bf16)
        | (1u << 20)    // pad_enable
        | (3u << 22)    // pad_interval: 16 DWORDs
        | (3u << 25);   // pad_amount:   4 DWORDs (16B)
  g1[1] = (tensor_d0 & 0xFFFFu) << 16;
  g1[2] = (tensor_d0 >> 16) | ((tensor_d1 & 0xFFFFu) << 16);
  g1[3] = (tensor_d1 >> 16) | (tile_d0 << 16);
  g1[4] = tile_d1;
  g1[5] = stride0;
  g1[6] = 0u;
  g1[7] = 0u;
  asm volatile("tensor_load_to_lds %0, %1" :: "s"(g0), "s"(g1) : "memory");
}

// ---------------------------------------------------------------------------
// Weight prep: W[K,N] f32 -> Wt[N,K] bf16 (32x32 tiles through LDS)
// ---------------------------------------------------------------------------
__global__ __launch_bounds__(256) void transpose_bf16(
    const float* __restrict__ W, unsigned short* __restrict__ Wt, int K, int N) {
  __shared__ __attribute__((aligned(16))) unsigned short t[32][33];
  const int n0 = blockIdx.x * 32;
  const int k0 = blockIdx.y * 32;
  const int c  = threadIdx.x & 31;
  const int r0 = threadIdx.x >> 5;
#pragma unroll
  for (int rr = 0; rr < 32; rr += 8)
    t[r0 + rr][c] = f2bf_bits(W[(size_t)(k0 + r0 + rr) * N + n0 + c]);
  __syncthreads();
#pragma unroll
  for (int rr = 0; rr < 32; rr += 8)
    Wt[(size_t)(n0 + r0 + rr) * K + k0 + c] = t[c][r0 + rr];
}

// ---------------------------------------------------------------------------
// LayerNorm: fp32 in -> bf16 out; one 256-thread block per row
// ---------------------------------------------------------------------------
__global__ __launch_bounds__(256) void ln_kernel(const float* __restrict__ x,
                                                 const float* __restrict__ g,
                                                 const float* __restrict__ b,
                                                 unsigned short* __restrict__ out) {
  __shared__ float rsum[8], rsq[8];
  const int row = blockIdx.x;
  const float* xr = x + (size_t)row * D_MODEL;
  float sum = 0.f, sq = 0.f;
  for (int i = threadIdx.x; i < D_MODEL; i += 256) {
    float v = xr[i]; sum += v; sq += v * v;
  }
#pragma unroll
  for (int o = 16; o > 0; o >>= 1) {
    sum += __shfl_xor(sum, o, 32);
    sq  += __shfl_xor(sq,  o, 32);
  }
  if ((threadIdx.x & 31) == 0) { rsum[threadIdx.x >> 5] = sum; rsq[threadIdx.x >> 5] = sq; }
  __syncthreads();
  sum = 0.f; sq = 0.f;
#pragma unroll
  for (int i = 0; i < 8; ++i) { sum += rsum[i]; sq += rsq[i]; }
  const float mu  = sum * (1.0f / D_MODEL);
  const float var = sq * (1.0f / D_MODEL) - mu * mu;
  const float rs  = rsqrtf(var + 1e-5f);
  unsigned short* orow = out + (size_t)row * D_MODEL;
  for (int i = threadIdx.x; i < D_MODEL; i += 256)
    orow[i] = f2bf_bits((xr[i] - mu) * rs * g[i] + b[i]);
}

// ---------------------------------------------------------------------------
// bf16 WMMA GEMM, distributed-TDM staged, double-buffered LDS.
//   C[M,N] = epi(A[M,K](bf16) @ Bt[N,K](bf16)^T + bias)
//   Workgroup tile 128x256, K step 32. 8 waves (2x4), each wave owns 64x64
//   of C as 4x4 WMMA tiles: 16 WMMA per 16 ds_load_b128 per K step.
//   Each wave DMAs its own slice (16 rows of A, 32 rows of B) via TDM,
//   waits its own TENSORcnt, one barrier per K step publishes the tile.
//   Epilogues: +bias [,GELU] [,+residual f32]; out f32 / bf16 / bf16
//   transposed to [B,H,dk,S] (V projection, vectorized 16B stores).
// ---------------------------------------------------------------------------
#define TM 128
#define TN 256
#define TK 32
#define LDK 40   // bf16 elements per LDS row (80B, matches TDM pad)

template <bool GELU, bool RES, bool OUTF32, bool TRANSV>
__global__ __launch_bounds__(256) void gemm_tdm(
    const unsigned short* __restrict__ A,   // bf16 [M,K]
    const unsigned short* __restrict__ Bt,  // bf16 [N,K]
    const float* __restrict__ bias,
    const float* __restrict__ residual,     // f32 [M,N] (if RES)
    float* __restrict__ Cf,                 // f32 out (if OUTF32)
    unsigned short* __restrict__ Cb,        // bf16 out (if !OUTF32)
    int M, int N, int K) {
  __shared__ __attribute__((aligned(16))) unsigned short As[2][TM * LDK];
  __shared__ __attribute__((aligned(16))) unsigned short Bs[2][TN * LDK];

  const int tid  = threadIdx.x;
  const int lane = tid & 31;
  const int wave = tid >> 5;     // 0..7
  const int wm   = wave & 1;     // 2 waves along M (64 rows each)
  const int wn   = wave >> 1;    // 4 waves along N (64 cols each)
  const int m0   = blockIdx.y * TM;
  const int n0   = blockIdx.x * TN;

  const unsigned ldsA[2] = { (unsigned)(uintptr_t)&As[0][0], (unsigned)(uintptr_t)&As[1][0] };
  const unsigned ldsB[2] = { (unsigned)(uintptr_t)&Bs[0][0], (unsigned)(uintptr_t)&Bs[1][0] };
  const unsigned aSlice = (unsigned)wave * 16u * (LDK * 2);
  const unsigned bSlice = (unsigned)wave * 32u * (LDK * 2);
  const size_t  aRow = (size_t)(m0 + wave * 16) * K;
  const size_t  bRow = (size_t)(n0 + wave * 32) * K;

  v8f acc[4][4];
#pragma unroll
  for (int i = 0; i < 4; ++i)
#pragma unroll
    for (int j = 0; j < 4; ++j)
#pragma unroll
      for (int r = 0; r < 8; ++r) acc[i][j][r] = 0.f;

  const int fm = lane & 15;
  const int fk = (lane >> 4) * 8;

  const int nsteps = K / TK;
  tdm_load_2d(ldsA[0] + aSlice, A + aRow, K, M, TK, 16, K);
  tdm_load_2d(ldsB[0] + bSlice, Bt + bRow, K, N, TK, 32, K);
  __builtin_amdgcn_s_wait_tensorcnt(0);
  __syncthreads();

  for (int s = 0; s < nsteps; ++s) {
    const int cur = s & 1;
    const bool pf = (s + 1 < nsteps);
    if (pf) {   // prefetch next K tile into the other buffer (overlaps WMMA)
      const int kk = (s + 1) * TK;
      tdm_load_2d(ldsA[1 - cur] + aSlice, A + aRow + kk, K, M, TK, 16, K);
      tdm_load_2d(ldsB[1 - cur] + bSlice, Bt + bRow + kk, K, N, TK, 32, K);
    }

    const unsigned short* as = As[cur];
    const unsigned short* bs = Bs[cur];
    v16bf bfrag[4];
#pragma unroll
    for (int j = 0; j < 4; ++j)
      bfrag[j] = load16(&bs[(wn * 64 + j * 16 + fm) * LDK + fk]);
#pragma unroll
    for (int i = 0; i < 4; ++i) {
      v16bf afrag = load16(&as[(wm * 64 + i * 16 + fm) * LDK + fk]);
#pragma unroll
      for (int j = 0; j < 4; ++j)
        acc[i][j] = wmma_bf16(afrag, bfrag[j], acc[i][j]);
    }

    if (pf) __builtin_amdgcn_s_wait_tensorcnt(0);   // own slice landed
    __syncthreads();                                // whole tile published
  }

  const int rofs = (lane >> 4) * 8;
  if (TRANSV) {
    // V projection: store bf16 transposed to [B, H, dk, S]; each lane's 8
    // accumulator rows are 8 consecutive s -> one 16B store per tile.
    const int bidx  = m0 / SEQN;            // whole block within one batch
    const int srow0 = m0 - bidx * SEQN + wm * 64 + rofs;
#pragma unroll
    for (int i = 0; i < 4; ++i) {
#pragma unroll
      for (int j = 0; j < 4; ++j) {
        const int gn = n0 + wn * 64 + j * 16 + fm;
        const float bv = bias[gn];
        const int hh = gn >> 7;             // head
        const int dd = gn & (DKH - 1);      // dim within head
        v8us pk;
#pragma unroll
        for (int r = 0; r < 8; ++r) pk[r] = f2bf_bits(acc[i][j][r] + bv);
        *(v8us*)&Cb[((size_t)(bidx * NHEAD + hh) * DKH + dd) * SEQN
                    + srow0 + i * 16] = pk;
      }
    }
  } else {
#pragma unroll
    for (int i = 0; i < 4; ++i) {
#pragma unroll
      for (int j = 0; j < 4; ++j) {
        const int gn = n0 + wn * 64 + j * 16 + fm;
        const float bv = bias[gn];
#pragma unroll
        for (int r = 0; r < 8; ++r) {
          const int gm = m0 + wm * 64 + i * 16 + rofs + r;
          float v = acc[i][j][r] + bv;
          if (GELU) v = 0.5f * v * (1.0f + erff(v * 0.70710678118654752f));
          if (RES) v += residual[(size_t)gm * N + gn];
          if (OUTF32) Cf[(size_t)gm * N + gn] = v;
          else        Cb[(size_t)gm * N + gn] = f2bf_bits(v);
        }
      }
    }
  }
}

// ---------------------------------------------------------------------------
// Flash attention, causal, bf16. One wave per (b, h, 16-query tile).
// Q/K: [B,S,H*dk]; V: [B,H,dk,S] (pre-transposed by the V projection).
// All fragments preloaded before WMMA bursts so loads overlap matrix ops.
// ---------------------------------------------------------------------------
__global__ __launch_bounds__(32) void attn_kernel(
    const unsigned short* __restrict__ Qp, const unsigned short* __restrict__ Kp,
    const unsigned short* __restrict__ Vt, unsigned short* __restrict__ Op) {
  __shared__ __attribute__((aligned(16))) unsigned short Ps[16 * LDK];

  const int lane = threadIdx.x & 31;
  const int fm   = lane & 15;
  const int hf   = lane >> 4;
  const int fk   = hf * 8;

  const int q0 = blockIdx.x * 16;
  const int bh = blockIdx.y;
  const int b  = bh / NHEAD;
  const int h  = bh % NHEAD;
  const int D  = NHEAD * DKH;
  const size_t base  = ((size_t)b * SEQN) * D + (size_t)h * DKH;
  const size_t vbase = ((size_t)bh * DKH) * SEQN;       // V^T head slab

  v16bf qf[4];
  {
    const unsigned short* qrow = Qp + base + (size_t)(q0 + fm) * D;
#pragma unroll
    for (int c = 0; c < 4; ++c) qf[c] = load16(qrow + c * 32 + fk);
  }

  v8f o_acc[8];
#pragma unroll
  for (int c = 0; c < 8; ++c)
#pragma unroll
    for (int r = 0; r < 8; ++r) o_acc[c][r] = 0.f;
  float mi[8], li[8];
#pragma unroll
  for (int r = 0; r < 8; ++r) { mi[r] = -1e30f; li[r] = 0.f; }

  const float scale = 0.0883883476483184f;  // 1/sqrt(128)
  const int nblk = (q0 + 16 + 31) / 32;

  for (int blk = 0; blk < nblk; ++blk) {
    const int sk = blk * 32;

    // ---- preload all 8 K fragments, then 8 WMMA (loads overlap compute) ----
    v16bf kf[8];
    {
      const unsigned short* k0 = Kp + base + (size_t)(sk + fm) * D;
      const unsigned short* k1 = k0 + (size_t)16 * D;
#pragma unroll
      for (int c = 0; c < 4; ++c) {
        kf[c]     = load16(k0 + c * 32 + fk);
        kf[4 + c] = load16(k1 + c * 32 + fk);
      }
    }
    v8f s0, s1;
#pragma unroll
    for (int r = 0; r < 8; ++r) { s0[r] = 0.f; s1[r] = 0.f; }
#pragma unroll
    for (int c = 0; c < 4; ++c) {
      s0 = wmma_bf16(qf[c], kf[c], s0);
      s1 = wmma_bf16(qf[c], kf[4 + c], s1);
    }

    // ---- online softmax ----
#pragma unroll
    for (int r = 0; r < 8; ++r) {
      const int qi = q0 + hf * 8 + r;
      float v0 = s0[r] * scale;
      float v1 = s1[r] * scale;
      if (sk + fm > qi)      v0 = -1e30f;
      if (sk + 16 + fm > qi) v1 = -1e30f;
      float mx = fmaxf(v0, v1);
#pragma unroll
      for (int o = 8; o > 0; o >>= 1) mx = fmaxf(mx, __shfl_xor(mx, o, 32));
      const float mnew  = fmaxf(mi[r], mx);
      const float alpha = __expf(mi[r] - mnew);
      const float p0 = __expf(v0 - mnew);
      const float p1 = __expf(v1 - mnew);
      float rs = p0 + p1;
#pragma unroll
      for (int o = 8; o > 0; o >>= 1) rs += __shfl_xor(rs, o, 32);
      li[r] = li[r] * alpha + rs;
      mi[r] = mnew;
#pragma unroll
      for (int c = 0; c < 8; ++c) o_acc[c][r] *= alpha;
      const int row = r + 8 * hf;
      Ps[row * LDK + fm]      = f2bf_bits(p0);
      Ps[row * LDK + 16 + fm] = f2bf_bits(p1);
    }
    __syncthreads();

    v16bf pf = load16(&Ps[fm * LDK + fk]);

    // ---- preload all 8 V fragments (contiguous: V is [dk][S]), 8 WMMA ----
    v16bf vf[8];
#pragma unroll
    for (int c = 0; c < 8; ++c)
      vf[c] = load16(Vt + vbase + (size_t)(c * 16 + fm) * SEQN + sk + fk);
#pragma unroll
    for (int c = 0; c < 8; ++c)
      o_acc[c] = wmma_bf16(pf, vf[c], o_acc[c]);
    __syncthreads();
  }

#pragma unroll
  for (int r = 0; r < 8; ++r) {
    const int qi = q0 + hf * 8 + r;
    const float inv = 1.0f / li[r];
#pragma unroll
    for (int c = 0; c < 8; ++c)
      Op[base + (size_t)qi * D + c * 16 + fm] = f2bf_bits(o_acc[c][r] * inv);
  }
}

// ---------------------------------------------------------------------------
// Host orchestration (graph-capture safe)
// ---------------------------------------------------------------------------
extern "C" void kernel_launch(void* const* d_in, const int* in_sizes, int n_in,
                              void* d_out, int out_size, void* d_ws, size_t ws_size,
                              hipStream_t stream) {
  (void)in_sizes; (void)n_in; (void)out_size; (void)ws_size;
  const float* x    = (const float*)d_in[0];
  const float* Wq   = (const float*)d_in[1];
  const float* bq   = (const float*)d_in[2];
  const float* Wk   = (const float*)d_in[3];
  const float* bk   = (const float*)d_in[4];
  const float* Wv   = (const float*)d_in[5];
  const float* bv   = (const float*)d_in[6];
  const float* Wo   = (const float*)d_in[7];
  const float* bo   = (const float*)d_in[8];
  const float* W1   = (const float*)d_in[9];
  const float* b1   = (const float*)d_in[10];
  const float* W2   = (const float*)d_in[11];
  const float* b2   = (const float*)d_in[12];
  const float* ln1g = (const float*)d_in[13];
  const float* ln1b = (const float*)d_in[14];
  const float* ln2g = (const float*)d_in[15];
  const float* ln2b = (const float*)d_in[16];
  float* out = (float*)d_out;

  const size_t MD = (size_t)MTOT * D_MODEL;
  const size_t DD = (size_t)D_MODEL * D_MODEL;
  const size_t DF = (size_t)D_MODEL * D_FF;
  char* wsb = (char*)d_ws;
  size_t off = 0;
  unsigned short* hb  = (unsigned short*)(wsb + off); off += MD * 2;   // LN1 out / attn out
  unsigned short* qb  = (unsigned short*)(wsb + off); off += MD * 2;   // Q / LN2 out
  unsigned short* kb  = (unsigned short*)(wsb + off); off += MD * 2;
  unsigned short* vb  = (unsigned short*)(wsb + off); off += MD * 2;   // V^T [B,H,dk,S]
  unsigned short* g1b = (unsigned short*)(wsb + off); off += (size_t)MTOT * D_FF * 2;
  float*          xab = (float*)(wsb + off);          off += MD * 4;   // x + attn out
  unsigned short* wqt = (unsigned short*)(wsb + off); off += DD * 2;
  unsigned short* wkt = (unsigned short*)(wsb + off); off += DD * 2;
  unsigned short* wvt = (unsigned short*)(wsb + off); off += DD * 2;
  unsigned short* wot = (unsigned short*)(wsb + off); off += DD * 2;
  unsigned short* w1t = (unsigned short*)(wsb + off); off += DF * 2;
  unsigned short* w2t = (unsigned short*)(wsb + off); off += DF * 2;

  const dim3 blk256(256), blk32(32);
  const dim3 trDD(D_MODEL / 32, D_MODEL / 32);
  const dim3 trW1(D_FF / 32,    D_MODEL / 32);
  const dim3 trW2(D_MODEL / 32, D_FF / 32);
  const dim3 gemmD(D_MODEL / TN, MTOT / TM);     // (8, 64)
  const dim3 gemmF(D_FF   / TN, MTOT / TM);      // (32, 64)
  const dim3 attnG(SEQN / 16, BATCHN * NHEAD);   // (128, 64)

  // one-time weight prep: fp32 [K,N] -> bf16 [N,K]
  transpose_bf16<<<trDD, blk256, 0, stream>>>(Wq, wqt, D_MODEL, D_MODEL);
  transpose_bf16<<<trDD, blk256, 0, stream>>>(Wk, wkt, D_MODEL, D_MODEL);
  transpose_bf16<<<trDD, blk256, 0, stream>>>(Wv, wvt, D_MODEL, D_MODEL);
  transpose_bf16<<<trDD, blk256, 0, stream>>>(Wo, wot, D_MODEL, D_MODEL);
  transpose_bf16<<<trW1, blk256, 0, stream>>>(W1, w1t, D_MODEL, D_FF);
  transpose_bf16<<<trW2, blk256, 0, stream>>>(W2, w2t, D_FF, D_MODEL);

  // h = LN1(x)
  ln_kernel<<<MTOT, blk256, 0, stream>>>(x, ln1g, ln1b, hb);
  // Q/K projections (bf16 out), V projection (bf16 out, transposed [B,H,dk,S])
  gemm_tdm<false,false,false,false><<<gemmD, blk256, 0, stream>>>(hb, wqt, bq, nullptr, nullptr, qb, MTOT, D_MODEL, D_MODEL);
  gemm_tdm<false,false,false,false><<<gemmD, blk256, 0, stream>>>(hb, wkt, bk, nullptr, nullptr, kb, MTOT, D_MODEL, D_MODEL);
  gemm_tdm<false,false,false,true ><<<gemmD, blk256, 0, stream>>>(hb, wvt, bv, nullptr, nullptr, vb, MTOT, D_MODEL, D_MODEL);
  // causal flash MHA (bf16 out into hb)
  attn_kernel<<<attnG, blk32, 0, stream>>>(qb, kb, vb, hb);
  // xa = x + attn @ Wo + bo   (f32 out)
  gemm_tdm<false,true,true,false><<<gemmD, blk256, 0, stream>>>(hb, wot, bo, x, xab, nullptr, MTOT, D_MODEL, D_MODEL);
  // h2 = LN2(xa) (bf16, reuse qb)
  ln_kernel<<<MTOT, blk256, 0, stream>>>(xab, ln2g, ln2b, qb);
  // g1 = gelu(h2 @ W1 + b1)  (bf16 out)
  gemm_tdm<true,false,false,false><<<gemmF, blk256, 0, stream>>>(qb, w1t, b1, nullptr, nullptr, g1b, MTOT, D_FF, D_MODEL);
  // out = xa + g1 @ W2 + b2  (f32 out)
  gemm_tdm<false,true,true,false><<<gemmD, blk256, 0, stream>>>(g1b, w2t, b2, xab, out, nullptr, MTOT, D_MODEL, D_FF);
}
